// GatedSparseAttention_40535901340240
// MI455X (gfx1250) — matrix operand order repaired and checked by reading, not compiled, not run on hardware
//
#include <hip/hip_runtime.h>
#include <hip/hip_bf16.h>
#include <math.h>

// ---------------------------------------------------------------------------
// GatedSparseAttention for MI455X (gfx1250, wave32, WMMA, async-LDS copies)
// ---------------------------------------------------------------------------

#define T_DIM 2048
#define D_DIM 512
#define H_DIM 8
#define DH_DIM 64
#define NIDX 4
#define DIDX 64
#define KSEL 128

// Flip to 0 if the assembler rejects the async mnemonic.
#define USE_ASYNC_LDS 1

typedef __attribute__((ext_vector_type(16))) __bf16 v16bf;
typedef __attribute__((ext_vector_type(8)))  float  v8f;

__device__ __forceinline__ unsigned short f2bf(float f) {
    unsigned u = __float_as_uint(f);
    unsigned r = (u + 0x7FFFu + ((u >> 16) & 1u)) >> 16;
    return (unsigned short)r;
}

union FragBF { v16bf v; uint4 u[2]; };

// A-fragment (16x32 bf16) per ISA 7.12.2: lanes<16 hold K {0..7}∪{16..23},
// lanes>=16 hold K {8..15}∪{24..31}.
__device__ __forceinline__ v16bf load_fragA(const unsigned short* rowp, int hh) {
    FragBF f;
    f.u[0] = *(const uint4*)(rowp + hh * 8);
    f.u[1] = *(const uint4*)(rowp + 16 + hh * 8);
    return f.v;
}
// B-fragment (32x16 bf16) from row-major N x K source: lane holds column
// n = lane&15; VGPRs 0..3 = K hh*8+0..7, VGPRs 4..7 = K 16+hh*8+0..7.
__device__ __forceinline__ v16bf load_fragB(const unsigned short* rowp, int hh) {
    FragBF f;
    f.u[0] = *(const uint4*)(rowp + hh * 8);
    f.u[1] = *(const uint4*)(rowp + 16 + hh * 8);
    return f.v;
}

// 16B global -> LDS copy through the async path (ASYNCcnt-tracked), with a
// synchronous fallback. LDS operand is the wave-relative 32-bit LDS address
// (low 32 bits of the flat pointer).
__device__ __forceinline__ void copy16_to_lds(const unsigned short* g, unsigned short* l) {
#if USE_ASYNC_LDS
    unsigned lds_addr = (unsigned)(uintptr_t)l;
    asm volatile("global_load_async_to_lds_b128 %0, %1, off"
                 :: "v"(lds_addr), "v"(g) : "memory");
#else
    *(uint4*)l = *(const uint4*)g;
#endif
}
__device__ __forceinline__ void wait_lds_copies() {
#if USE_ASYNC_LDS
    asm volatile("s_wait_asynccnt 0x0" ::: "memory");
#endif
}

// ------------------------- conversion kernels ------------------------------
__global__ void cvt_bf16_kernel(const float* __restrict__ in,
                                unsigned short* __restrict__ out, int n) {
    int i = blockIdx.x * blockDim.x + threadIdx.x;
    if (i < n) out[i] = f2bf(in[i]);
}

// in: K x N row-major (fp32 weight) -> out: N x K row-major bf16 (transposed)
__global__ void cvt_bf16_t_kernel(const float* __restrict__ in,
                                  unsigned short* __restrict__ out, int K, int N) {
    int i = blockIdx.x * blockDim.x + threadIdx.x;
    if (i < K * N) {
        int k = i / N, n = i - k * N;
        out[(size_t)n * K + k] = f2bf(in[i]);
    }
}

// ------------------------- LDS-blocked WMMA GEMM ---------------------------
// C[M x N] = A[M x K] @ Bt[N x K]^T ; mode 1 => C = sigmoid(C + bias[n])
// Block tile 64x64, 8 waves (4M x 2N), each wave 16x32 (2 WMMA accumulators).
#define BM 64
#define BN 64
#define BK 32
__global__ __launch_bounds__(256) void gemm_bf16_lds(
    const unsigned short* __restrict__ A, const unsigned short* __restrict__ Bt,
    float* __restrict__ C, const float* __restrict__ bias,
    int M, int N, int K, int mode) {
    __shared__ __attribute__((aligned(16))) unsigned short As[BM * BK];
    __shared__ __attribute__((aligned(16))) unsigned short Bs[BN * BK];

    int tid = threadIdx.x;
    int wid = tid >> 5, lane = tid & 31;
    int wm = wid >> 1, wn = wid & 1;           // 4 x 2 wave grid
    int nbx = N / BN;
    int bx = blockIdx.x % nbx;
    int by = blockIdx.x / nbx;
    int hh = lane >> 4, rm = lane & 15;

    // copy mapping: 256 threads, each moves one 16B chunk of A and one of B
    int crow = tid >> 2, cchk = (tid & 3) * 8;  // row 0..63, K-chunk offset
    const unsigned short* ag = A  + (size_t)(by * BM + crow) * K + cchk;
    const unsigned short* bg = Bt + (size_t)(bx * BN + crow) * K + cchk;
    unsigned short* al = &As[crow * BK + cchk];
    unsigned short* bl = &Bs[crow * BK + cchk];

    const unsigned short* afrag = &As[(wm * 16 + rm) * BK];
    const unsigned short* bfrag0 = &Bs[(wn * 32 + rm) * BK];
    const unsigned short* bfrag1 = &Bs[(wn * 32 + 16 + rm) * BK];

    v8f acc0 = {}, acc1 = {};
    for (int k0 = 0; k0 < K; k0 += BK) {
        copy16_to_lds(ag + k0, al);
        copy16_to_lds(bg + k0, bl);
        if (k0 + BK < K) {                      // prefetch next K-step
            __builtin_prefetch(ag + k0 + BK, 0, 0);
            __builtin_prefetch(bg + k0 + BK, 0, 0);
        }
        wait_lds_copies();
        __syncthreads();
        v16bf a  = load_fragA(afrag, hh);
        v16bf b0 = load_fragB(bfrag0, hh);
        v16bf b1 = load_fragB(bfrag1, hh);
        acc0 = __builtin_amdgcn_wmma_f32_16x16x32_bf16(
            false, a, false, b0, (short)0, acc0, false, false);
        acc1 = __builtin_amdgcn_wmma_f32_16x16x32_bf16(
            false, a, false, b1, (short)0, acc1, false, false);
        __syncthreads();
    }

    int row0 = by * BM + wm * 16 + hh * 8;
    int col0 = bx * BN + wn * 32 + rm;
    if (mode == 0) {
#pragma unroll
        for (int r = 0; r < 8; ++r) {
            C[(size_t)(row0 + r) * N + col0]      = acc0[r];
            C[(size_t)(row0 + r) * N + col0 + 16] = acc1[r];
        }
    } else {
        float bb0 = bias ? bias[col0] : 0.f;
        float bb1 = bias ? bias[col0 + 16] : 0.f;
#pragma unroll
        for (int r = 0; r < 8; ++r) {
            float s0 = acc0[r] + bb0, s1 = acc1[r] + bb1;
            C[(size_t)(row0 + r) * N + col0]      = 1.f / (1.f + __expf(-s0));
            C[(size_t)(row0 + r) * N + col0 + 16] = 1.f / (1.f + __expf(-s1));
        }
    }
}

// -------------------- index mixing weights: sigmoid(x@Wiw+biw) -------------
__global__ __launch_bounds__(128) void index_w_kernel(
    const float* __restrict__ x, const float* __restrict__ Wiw,
    const float* __restrict__ biw, float* __restrict__ wsig, int T, int D) {
    int wid = blockIdx.x * (blockDim.x >> 5) + (threadIdx.x >> 5);
    int lane = threadIdx.x & 31;
    int t = wid >> 2, h = wid & 3;
    if (t >= T) return;
    float s = 0.f;
    for (int d = lane; d < D; d += 32) s += x[(size_t)t * D + d] * Wiw[d * NIDX + h];
#pragma unroll
    for (int off = 16; off; off >>= 1) s += __shfl_xor(s, off, 32);
    if (lane == 0) wsig[t * NIDX + h] = 1.f / (1.f + __expf(-(s + biw[h])));
}

// ------------------------------- elementwise -------------------------------
__global__ void vgate_kernel(float* __restrict__ v, const float* __restrict__ g, int n) {
    int i = blockIdx.x * blockDim.x + threadIdx.x;
    if (i < n) v[i] *= g[i];
}

__global__ void rope_kernel(float* __restrict__ q, int T) {
    const int hd2 = DH_DIM / 2;
    int i = blockIdx.x * blockDim.x + threadIdx.x;       // T*H*hd2 threads
    if (i >= T * H_DIM * hd2) return;
    int d = i % hd2;
    int th = i / hd2;
    int h = th % H_DIM, t = th / H_DIM;
    float inv_freq = __expf(-((float)(2 * d) / DH_DIM) * 9.210340371976184f); // ln(1e4)
    float ang = (float)t * inv_freq;
    float c, s;
    __sincosf(ang, &s, &c);
    float* p = q + (size_t)t * D_DIM + h * DH_DIM;
    float a = p[d], b = p[d + hd2];
    p[d]       = a * c - b * s;
    p[d + hd2] = b * c + a * s;
}

// -------------------- fused 4-head WMMA index scoring ----------------------
// S[q][kk] = sum_h sigmoid(qi_h . ki / 8 + idx_bias[h]) * wsig[q][h]; causal -inf
__global__ __launch_bounds__(256) void index_scores_wmma(
    const unsigned short* __restrict__ QIbf,   // T x (NIDX*DIDX)
    const unsigned short* __restrict__ KIbf,   // T x DIDX
    const float* __restrict__ wsig,            // T x NIDX (already sigmoid)
    const float* __restrict__ idx_bias,        // NIDX
    float* __restrict__ S, int T) {
    int wave = blockIdx.x * (blockDim.x >> 5) + (threadIdx.x >> 5);
    int lane = threadIdx.x & 31;
    int ntx = T >> 4;
    int tk = wave % ntx;
    int tq = wave / ntx;
    if (tq >= (T >> 4)) return;
    int hh = lane >> 4, rm = lane & 15;

    const unsigned short* brow = KIbf + (size_t)(tk * 16 + rm) * DIDX;
    v16bf b0 = load_fragB(brow, hh);
    v16bf b1 = load_fragB(brow + 32, hh);
    v8f acc[NIDX] = {};
#pragma unroll
    for (int h = 0; h < NIDX; ++h) {
        const unsigned short* arow =
            QIbf + (size_t)(tq * 16 + rm) * (NIDX * DIDX) + h * DIDX;
        v16bf a0 = load_fragA(arow, hh);
        v16bf a1 = load_fragA(arow + 32, hh);
        acc[h] = __builtin_amdgcn_wmma_f32_16x16x32_bf16(
            false, a0, false, b0, (short)0, acc[h], false, false);
        acc[h] = __builtin_amdgcn_wmma_f32_16x16x32_bf16(
            false, a1, false, b1, (short)0, acc[h], false, false);
    }
    int col = tk * 16 + rm;
    const float inv = 0.125f;                   // 1/sqrt(DIDX)
#pragma unroll
    for (int r = 0; r < 8; ++r) {
        int row = tq * 16 + hh * 8 + r;
        float s = 0.f;
#pragma unroll
        for (int h = 0; h < NIDX; ++h) {
            float g = 1.f / (1.f + __expf(-(acc[h][r] * inv + idx_bias[h])));
            s += g * wsig[row * NIDX + h];
        }
        S[(size_t)row * T + col] = (col > row) ? -INFINITY : s;
    }
}

// ------------------------------- top-k (stable) ----------------------------
__global__ __launch_bounds__(256) void topk_kernel(
    const float* __restrict__ S, int* __restrict__ idxout, int T) {
    __shared__ float row[T_DIM];
    __shared__ float rmax[256];
    __shared__ int   rarg[256];
    int q = blockIdx.x;
    const float* s = S + (size_t)q * T;
    for (int j = threadIdx.x; j < T; j += 256) {
        float v = s[j];
        row[j] = isinf(v) ? -1e9f : v;          // matches scores_for_topk
    }
    __syncthreads();
    for (int sel = 0; sel < KSEL; ++sel) {
        float best = -INFINITY;
        int barg = T;
        for (int j = threadIdx.x; j < T; j += 256) {
            float v = row[j];
            if (v > best) { best = v; barg = j; }
        }
        rmax[threadIdx.x] = best;
        rarg[threadIdx.x] = barg;
        __syncthreads();
        for (int off = 128; off; off >>= 1) {
            if (threadIdx.x < off) {
                float o = rmax[threadIdx.x + off];
                int  oa = rarg[threadIdx.x + off];
                if (o > rmax[threadIdx.x] ||
                    (o == rmax[threadIdx.x] && oa < rarg[threadIdx.x])) {
                    rmax[threadIdx.x] = o;
                    rarg[threadIdx.x] = oa;
                }
            }
            __syncthreads();
        }
        if (threadIdx.x == 0) {
            idxout[(size_t)q * KSEL + sel] = rarg[0];
            row[rarg[0]] = -INFINITY;
        }
        __syncthreads();
    }
}

// --------------- gathered 128-key attention, one wave per (q,h) ------------
__global__ __launch_bounds__(128) void attn_kernel(
    const float* __restrict__ q, const float* __restrict__ k,
    const float* __restrict__ v, const float* __restrict__ og,
    const float* __restrict__ S, const int* __restrict__ idx,
    float* __restrict__ out, int T) {
    __shared__ float qv[4][DH_DIM];
    __shared__ float pr[4][KSEL];
    __shared__ int   il[4][KSEL];
    int w = threadIdx.x >> 5, lane = threadIdx.x & 31;
    int gw = blockIdx.x * 4 + w;                // grid sized exactly: T*H/4
    int qrow = gw >> 3, h = gw & 7;

    const float* qp = q + (size_t)qrow * D_DIM + h * DH_DIM;
    qv[w][lane]      = qp[lane];
    qv[w][lane + 32] = qp[lane + 32];
    __syncthreads();

    const float scale = 0.125f;                 // dh^-0.5
    float sc[4];
#pragma unroll
    for (int c = 0; c < 4; ++c) {
        int j = c * 32 + lane;
        int key = idx[(size_t)qrow * KSEL + j];
        il[w][j] = key;
        bool m = !isinf(S[(size_t)qrow * T + key]);
        float d = 0.f;
        const float* kp = k + (size_t)key * D_DIM + h * DH_DIM;
#pragma unroll
        for (int dd = 0; dd < DH_DIM; dd += 4) {
            float4 kk = *(const float4*)(kp + dd);
            d += qv[w][dd] * kk.x + qv[w][dd + 1] * kk.y +
                 qv[w][dd + 2] * kk.z + qv[w][dd + 3] * kk.w;
        }
        sc[c] = m ? d * scale : -INFINITY;
    }
    float mx = fmaxf(fmaxf(sc[0], sc[1]), fmaxf(sc[2], sc[3]));
#pragma unroll
    for (int off = 16; off; off >>= 1) mx = fmaxf(mx, __shfl_xor(mx, off, 32));
    float e[4], Z = 0.f;
#pragma unroll
    for (int c = 0; c < 4; ++c) {
        e[c] = (isinf(sc[c]) || isinf(mx)) ? 0.f : __expf(sc[c] - mx);
        Z += e[c];
    }
#pragma unroll
    for (int off = 16; off; off >>= 1) Z += __shfl_xor(Z, off, 32);
    float invZ = (Z > 0.f) ? 1.f / Z : 0.f;
#pragma unroll
    for (int c = 0; c < 4; ++c) pr[w][c * 32 + lane] = e[c] * invZ;
    __syncthreads();

    int d0 = lane * 2;
    float a0 = 0.f, a1 = 0.f;
    for (int j = 0; j < KSEL; ++j) {
        float p = pr[w][j];
        if (p != 0.f) {
            const float* vp = v + (size_t)il[w][j] * D_DIM + h * DH_DIM;
            a0 += p * vp[d0];
            a1 += p * vp[d0 + 1];
        }
    }
    const float* gp = og + (size_t)qrow * D_DIM + h * DH_DIM;
    out[(size_t)qrow * D_DIM + h * DH_DIM + d0]     = a0 * gp[d0];
    out[(size_t)qrow * D_DIM + h * DH_DIM + d0 + 1] = a1 * gp[d0 + 1];
}

// ---------------------------------------------------------------------------
extern "C" void kernel_launch(void* const* d_in, const int* in_sizes, int n_in,
                              void* d_out, int out_size, void* d_ws, size_t ws_size,
                              hipStream_t stream) {
    (void)in_sizes; (void)n_in; (void)out_size; (void)ws_size;
    const int T = T_DIM, D = D_DIM;

    const float* x        = (const float*)d_in[0];
    const float* Wq       = (const float*)d_in[1];
    const float* Wk       = (const float*)d_in[2];
    const float* Wv       = (const float*)d_in[3];
    const float* Wo       = (const float*)d_in[4];
    const float* Wiq      = (const float*)d_in[5];
    const float* Wik      = (const float*)d_in[6];
    const float* Wiw      = (const float*)d_in[7];
    const float* biw      = (const float*)d_in[8];
    const float* idx_bias = (const float*)d_in[9];
    const float* Wvg      = (const float*)d_in[10];
    const float* bvg      = (const float*)d_in[11];
    const float* Wog      = (const float*)d_in[12];
    const float* bog      = (const float*)d_in[13];

    // ---- workspace carve-out (256B aligned) ----
    char* wsp = (char*)d_ws;
    auto carve = [&](size_t bytes) -> void* {
        void* p = (void*)wsp;
        wsp += (bytes + 255) & ~(size_t)255;
        return p;
    };
    unsigned short* xbf   = (unsigned short*)carve((size_t)T * D * 2);
    unsigned short* WqT   = (unsigned short*)carve((size_t)D * D * 2);
    unsigned short* WkT   = (unsigned short*)carve((size_t)D * D * 2);
    unsigned short* WvT   = (unsigned short*)carve((size_t)D * D * 2);
    unsigned short* WvgT  = (unsigned short*)carve((size_t)D * D * 2);
    unsigned short* WogT  = (unsigned short*)carve((size_t)D * D * 2);
    unsigned short* WoT   = (unsigned short*)carve((size_t)D * D * 2);
    unsigned short* WiqT  = (unsigned short*)carve((size_t)(NIDX * DIDX) * D * 2);
    unsigned short* WikT  = (unsigned short*)carve((size_t)DIDX * D * 2);
    float* qf   = (float*)carve((size_t)T * D * 4);
    float* kf   = (float*)carve((size_t)T * D * 4);
    float* vf   = (float*)carve((size_t)T * D * 4);
    float* vg   = (float*)carve((size_t)T * D * 4);
    float* og   = (float*)carve((size_t)T * D * 4);
    float* qi   = (float*)carve((size_t)T * NIDX * DIDX * 4);
    float* ki   = (float*)carve((size_t)T * DIDX * 4);
    unsigned short* qibf = (unsigned short*)carve((size_t)T * NIDX * DIDX * 2);
    unsigned short* kibf = (unsigned short*)carve((size_t)T * DIDX * 2);
    float* wsig = (float*)carve((size_t)T * NIDX * 4);
    float* S    = (float*)carve((size_t)T * T * 4);
    int*   sidx = (int*)carve((size_t)T * KSEL * 4);
    float* ao   = (float*)carve((size_t)T * D * 4);
    unsigned short* aobf = (unsigned short*)carve((size_t)T * D * 2);

    auto cvt = [&](const float* in, unsigned short* out, int n) {
        cvt_bf16_kernel<<<(n + 255) / 256, 256, 0, stream>>>(in, out, n);
    };
    auto cvtT = [&](const float* in, unsigned short* out, int K, int N) {
        cvt_bf16_t_kernel<<<(K * N + 255) / 256, 256, 0, stream>>>(in, out, K, N);
    };
    auto gemm = [&](const unsigned short* A, const unsigned short* Bt, float* C,
                    const float* bias, int M, int N, int K, int mode) {
        int blocks = (M / BM) * (N / BN);
        gemm_bf16_lds<<<blocks, 256, 0, stream>>>(A, Bt, C, bias, M, N, K, mode);
    };

    // 1) bf16 conversions: activations + transposed weights
    cvt(x, xbf, T * D);
    cvtT(Wq, WqT, D, D);
    cvtT(Wk, WkT, D, D);
    cvtT(Wv, WvT, D, D);
    cvtT(Wvg, WvgT, D, D);
    cvtT(Wog, WogT, D, D);
    cvtT(Wo, WoT, D, D);
    cvtT(Wiq, WiqT, D, NIDX * DIDX);
    cvtT(Wik, WikT, D, DIDX);

    // 2) projections (LDS-blocked WMMA); gates fuse sigmoid+bias
    gemm(xbf, WqT, qf, nullptr, T, D, D, 0);
    gemm(xbf, WkT, kf, nullptr, T, D, D, 0);
    gemm(xbf, WvT, vf, nullptr, T, D, D, 0);
    gemm(xbf, WvgT, vg, bvg, T, D, D, 1);
    gemm(xbf, WogT, og, bog, T, D, D, 1);
    gemm(xbf, WiqT, qi, nullptr, T, NIDX * DIDX, D, 0);
    gemm(xbf, WikT, ki, nullptr, T, DIDX, D, 0);

    // 3) index mixing weights (tiny N=4 GEMV, wave-reduced)
    index_w_kernel<<<(T * NIDX + 3) / 4, 128, 0, stream>>>(x, Wiw, biw, wsig, T, D);

    // 4) value gate + RoPE on q, k
    vgate_kernel<<<(T * D + 255) / 256, 256, 0, stream>>>(vf, vg, T * D);
    {
        int n = T * H_DIM * (DH_DIM / 2);
        rope_kernel<<<(n + 255) / 256, 256, 0, stream>>>(qf, T);
        rope_kernel<<<(n + 255) / 256, 256, 0, stream>>>(kf, T);
    }

    // 5) index scores (fused 4-head WMMA over 2048x2048)
    cvt(qi, qibf, T * NIDX * DIDX);
    cvt(ki, kibf, T * DIDX);
    {
        int tiles = (T / 16) * (T / 16);
        index_scores_wmma<<<(tiles + 7) / 8, 256, 0, stream>>>(
            qibf, kibf, wsig, idx_bias, S, T);
    }

    // 6) top-128 per query (stable)
    topk_kernel<<<T, 256, 0, stream>>>(S, sidx, T);

    // 7) gathered sparse attention + output gate
    attn_kernel<<<T * H_DIM / 4, 128, 0, stream>>>(qf, kf, vf, og, S, sidx, ao, T);

    // 8) output projection (WMMA) into d_out
    cvt(ao, aobf, T * D);
    gemm(aobf, WoT, (float*)d_out, nullptr, T, D, D, 0);
}
